// GeographyAwareAttention_13314398617911
// MI455X (gfx1250) — matrix operand order repaired and checked by reading, not compiled, not run on hardware
//
#include <hip/hip_runtime.h>
#include <math.h>

// Problem constants (match reference: B=4, T=1024, D=1024, H=16, DK=64)
#define B_  4
#define T_  1024
#define D_  1024
#define H_  16
#define DK_ 64

typedef __attribute__((ext_vector_type(16))) _Float16 v16h;
typedef __attribute__((ext_vector_type(8)))  _Float16 v8h;
typedef __attribute__((ext_vector_type(8)))  float    v8f;

typedef unsigned int u32x4 __attribute__((ext_vector_type(4)));
typedef int          i32x4 __attribute__((ext_vector_type(4)));
typedef int          i32x8 __attribute__((ext_vector_type(8)));

// ---------------------------------------------------------------------------
// Tensor Data Mover: 2-D tile load Global -> LDS (f16 elements).
// D# per CDNA5 ISA: group0 = {count/lds_addr/global_addr/type},
// group1 = {flags, tensor dims, tile dims, stride}. pad_enable inserts
// pad_amount DWORDs every pad_interval DWORDs -> reproduces padded LDS strides.
//   pad_interval_code: DWORDs before pad = 2 << code
//   pad_amount_code:   pad DWORDs        = code + 1
// ---------------------------------------------------------------------------
__device__ __forceinline__ void tdm_load_tile_2d(
    unsigned lds_off, const void* gaddr, int tile_w, int tile_h,
    int row_stride_elems, int pad_interval_code, int pad_amount_code) {
  unsigned long long ga = (unsigned long long)(size_t)gaddr;
  u32x4 g0;
  g0[0] = 1u;                                   // count = 1 valid descriptor
  g0[1] = lds_off;                              // LDS byte address
  g0[2] = (unsigned)ga;                         // global_addr[31:0]
  g0[3] = (unsigned)((ga >> 32) & 0x01ffffffull) | (2u << 30);  // [56:32]|type=2
  i32x8 g1 = {};
  g1[0] = (1 << 16)                             // data_size = 2 bytes
        | (1 << 20)                             // pad_enable
        | (pad_interval_code << 22)
        | (pad_amount_code << 25);
  g1[1] = (tile_w & 0xffff) << 16;              // tensor_dim0[15:0]
  g1[2] = (tile_h & 0xffff) << 16;              // tensor_dim1[15:0]
  g1[3] = (tile_w & 0xffff) << 16;              // tile_dim0
  g1[4] = (tile_h & 0xffff);                    // tile_dim1 (tile_dim2 = 0)
  g1[5] = row_stride_elems;                     // tensor_dim0_stride[31:0]
  i32x4 z4 = {};
#if defined(__clang_major__) && (__clang_major__ >= 23)
  i32x8 z8 = {};
  __builtin_amdgcn_tensor_load_to_lds(g0, g1, z4, z4, z8, 0);
#else
  __builtin_amdgcn_tensor_load_to_lds(g0, g1, z4, z4, 0);
#endif
}

__device__ __forceinline__ unsigned lds_offset_of(const void* p) {
  // generic LDS address = {shared aperture, lds byte offset} -> truncate
  return (unsigned)(size_t)p;
}

// ---------------------------------------------------------------------------
// CDNA5 WMMA fragment loaders (wave32), ISA 7.12.2 layouts.
// A (16x32 f16): lane L<16: row M=L, halfs A[M][0..7]++A[M][16..23];
//                lane>=16: same row, K ranges +8. Two b128 LDS reads.
// ---------------------------------------------------------------------------
__device__ __forceinline__ v16h ld_afrag(const _Float16* base, int stride) {
  const int lane = threadIdx.x & 31;
  const int m    = lane & 15;
  const int kh   = (lane >> 4) << 3;
  const _Float16* r = base + m * stride;
  v8h lo = *(const v8h*)(r + kh);
  v8h hi = *(const v8h*)(r + 16 + kh);
  return __builtin_shufflevector(lo, hi, 0,1,2,3,4,5,6,7,8,9,10,11,12,13,14,15);
}

// B (32x16 f16): lane L<16: col N=L holds K=0..7,16..23; lane>=16: K=8..15,24..31.
// Tile staged n-major (Bt[n][k]) -> two contiguous b128 LDS reads.
__device__ __forceinline__ v16h ld_bfrag(const _Float16* base, int stride) {
  const int lane = threadIdx.x & 31;
  const int n    = lane & 15;
  const int kh   = (lane >> 4) << 3;
  const _Float16* r = base + n * stride;
  v8h lo = *(const v8h*)(r + kh);
  v8h hi = *(const v8h*)(r + 16 + kh);
  return __builtin_shufflevector(lo, hi, 0,1,2,3,4,5,6,7,8,9,10,11,12,13,14,15);
}

// C/D (16x16 f32): VGPR r, lane L<16 -> (M=r, N=L); lane>=16 -> (M=r+8, N=L-16)

// ---------------------------------------------------------------------------
__global__ __launch_bounds__(256) void f32_to_f16_kernel(
    const float* __restrict__ in, _Float16* __restrict__ out, int n) {
  int i = blockIdx.x * 256 + threadIdx.x;
  if (i < n) out[i] = (_Float16)in[i];
}

// ---------------------------------------------------------------------------
// Tiled WMMA GEMM:  C[4096][1024] = A[4096][1024] @ W^T + bias
// Block tile 128x64, 8 waves (4x2), each wave 32x32 (2x2 wmma accums).
// Double-buffered LDS; tiles DMA'd by the Tensor Data Mover (wave 0 issues,
// TENSORcnt in-order completion lets next-tile DMAs stay in flight during
// the wait on the current pair).
// OUT_F16_QKV=1: f16 scaled into [B,H,T,DK]; =0: f32 row-major.
// ---------------------------------------------------------------------------
template <int OUT_F16_QKV>
__global__ __launch_bounds__(256) void gemm128x64(
    const _Float16* __restrict__ A, const _Float16* __restrict__ Bw,
    const float* __restrict__ bias, void* __restrict__ outp, float outScale) {
  constexpr int Kdim = D_;
  constexpr int NSTEP = Kdim / 32;
  __shared__ __align__(16) _Float16 As[2][128][40];  // 80B rows (pad 4 DW / 16 DW)
  __shared__ __align__(16) _Float16 Bs[2][64][40];

  const int tid  = threadIdx.x;
  const int wave = tid >> 5;
  const int wm   = wave >> 1, wn = wave & 1;
  const int bm   = blockIdx.y * 128, bn = blockIdx.x * 64;
  const bool issuer = (tid < 32);

  const unsigned asOff = lds_offset_of(&As[0][0][0]);
  const unsigned bsOff = lds_offset_of(&Bs[0][0][0]);
  constexpr unsigned asBufB = sizeof(_Float16) * 128 * 40;
  constexpr unsigned bsBufB = sizeof(_Float16) * 64 * 40;

  v8f acc[2][2] = {};

  if (issuer) {
    tdm_load_tile_2d(asOff, A + (size_t)bm * Kdim, 32, 128, Kdim, 3, 3);
    tdm_load_tile_2d(bsOff, Bw + (size_t)bn * Kdim, 32, 64, Kdim, 3, 3);
  }

  for (int s = 0; s < NSTEP; ++s) {
    const int cur = s & 1, nxt = cur ^ 1;
    if (issuer) {
      if (s + 1 < NSTEP) {
        const int k1 = (s + 1) * 32;
        tdm_load_tile_2d(asOff + nxt * asBufB, A + (size_t)bm * Kdim + k1,
                         32, 128, Kdim, 3, 3);
        tdm_load_tile_2d(bsOff + nxt * bsBufB, Bw + (size_t)bn * Kdim + k1,
                         32, 64, Kdim, 3, 3);
        __builtin_amdgcn_s_wait_tensorcnt(2);  // current pair done, next in flight
      } else {
        __builtin_amdgcn_s_wait_tensorcnt(0);
      }
    }
    __syncthreads();

    v16h af[2], bf[2];
#pragma unroll
    for (int tm = 0; tm < 2; ++tm)
      af[tm] = ld_afrag(&As[cur][wm * 32 + tm * 16][0], 40);
#pragma unroll
    for (int tn = 0; tn < 2; ++tn)
      bf[tn] = ld_bfrag(&Bs[cur][wn * 32 + tn * 16][0], 40);
#pragma unroll
    for (int tm = 0; tm < 2; ++tm)
#pragma unroll
      for (int tn = 0; tn < 2; ++tn)
        acc[tm][tn] = __builtin_amdgcn_wmma_f32_16x16x32_f16(
            false, af[tm], false, bf[tn], (short)0, acc[tm][tn], false, false);
    __syncthreads();
  }

  const int lane = tid & 31;
  const int cl = lane & 15, rh = (lane >> 4) << 3;
#pragma unroll
  for (int tm = 0; tm < 2; ++tm)
#pragma unroll
    for (int tn = 0; tn < 2; ++tn) {
      int gn = bn + wn * 32 + tn * 16 + cl;
      float bv = bias[gn];
#pragma unroll
      for (int r = 0; r < 8; ++r) {
        int gm = bm + wm * 32 + tm * 16 + r + rh;
        float v = acc[tm][tn][r] + bv;
        if (OUT_F16_QKV) {
          v *= outScale;  // Q gets 1/sqrt(DK) baked in here
          int b = gm >> 10, t = gm & (T_ - 1);
          int h = gn >> 6, dk = gn & (DK_ - 1);
          ((_Float16*)outp)[(((size_t)(b * H_ + h) * T_ + t) << 6) + dk] =
              (_Float16)v;
        } else {
          ((float*)outp)[(size_t)gm * D_ + gn] = v;
        }
      }
    }
}

// ---------------------------------------------------------------------------
// Flash attention with distance bias + causal mask.
// Grid: (B*H, T/64). 128 threads = 4 waves; wave w owns Q rows [16w,16w+16).
// Q/K tiles DMA'd by TDM (wave 0); V transposed manually by all lanes while
// the K DMA is in flight. Online softmax in C-layout registers.
// ---------------------------------------------------------------------------
__global__ __launch_bounds__(128) void attn_flash_kernel(
    const _Float16* __restrict__ Qh, const _Float16* __restrict__ Kh,
    const _Float16* __restrict__ Vh, const float* __restrict__ dist,
    const float* __restrict__ alphaP, _Float16* __restrict__ Oh) {
  __shared__ __align__(16) _Float16 Qs[64][72];  // q x dk (144B rows: 4DW/32DW pad)
  __shared__ __align__(16) _Float16 Ks[64][72];  // j x dk  (Bt[n=j][k=dk])
  __shared__ __align__(16) _Float16 Vt[64][72];  // dk x j  (Bt[n=dk][k=j])
  __shared__ __align__(16) _Float16 Ps[64][72];  // q x j

  const int tid = threadIdx.x, wave = tid >> 5, lane = tid & 31;
  const int bh = blockIdx.x, qt = blockIdx.y;
  const int b = bh >> 4, h = bh & (H_ - 1);
  const int q0 = qt * 64;
  const int cl = lane & 15, rh = (lane >> 4) << 3;
  const bool issuer = (tid < 32);
  const float alpha = fabsf(alphaP[0]);
  const size_t headBase = ((size_t)bh * T_) << 6;  // * DK

  // Q tile via Tensor Data Mover
  if (issuer) {
    tdm_load_tile_2d(lds_offset_of(&Qs[0][0]),
                     Qh + headBase + ((size_t)q0 << 6), 64, 64, DK_, 4, 3);
    __builtin_amdgcn_s_wait_tensorcnt(0);
  }
  __syncthreads();
  v16h qa0 = ld_afrag(&Qs[wave * 16][0], 72);   // K = dk 0..31
  v16h qa1 = ld_afrag(&Qs[wave * 16][32], 72);  // K = dk 32..63

  v8f aO[4] = {};
  float mrow[8], lrow[8];
#pragma unroll
  for (int r = 0; r < 8; ++r) { mrow[r] = -3.0e38f; lrow[r] = 0.0f; }

  for (int jb = 0; jb <= qt; ++jb) {
    const int j0 = jb * 64;
    // K tile DMA starts first ...
    if (issuer)
      tdm_load_tile_2d(lds_offset_of(&Ks[0][0]),
                       Kh + headBase + ((size_t)j0 << 6), 64, 64, DK_, 4, 3);
    // ... lanes transpose V into LDS while the DMA runs
#pragma unroll
    for (int p = 0; p < 4; ++p) {
      int idx = p * 128 + tid;
      int row = idx >> 3, cg = (idx & 7) << 3;
      v8h vv = *(const v8h*)(Vh + headBase + ((size_t)(j0 + row) << 6) + cg);
#pragma unroll
      for (int i = 0; i < 8; ++i) Vt[cg + i][row] = vv[i];
    }
    if (issuer) __builtin_amdgcn_s_wait_tensorcnt(0);
    __syncthreads();

    // S stripe (16 x 64) = Q K^T  -- hoist all B fragments, then wmma burst
    v16h kb0[4], kb1[4];
#pragma unroll
    for (int tn = 0; tn < 4; ++tn) {
      kb0[tn] = ld_bfrag(&Ks[tn * 16][0], 72);
      kb1[tn] = ld_bfrag(&Ks[tn * 16][32], 72);
    }
    float sv[4][8];
#pragma unroll
    for (int tn = 0; tn < 4; ++tn) {
      v8f z = {};
      z = __builtin_amdgcn_wmma_f32_16x16x32_f16(false, qa0, false, kb0[tn],
                                                 (short)0, z, false, false);
      z = __builtin_amdgcn_wmma_f32_16x16x32_f16(false, qa1, false, kb1[tn],
                                                 (short)0, z, false, false);
#pragma unroll
      for (int r = 0; r < 8; ++r) {
        int qg = q0 + wave * 16 + r + rh;
        int jg = j0 + tn * 16 + cl;
        float v = z[r] - alpha * dist[((size_t)(b * T_ + qg)) * T_ + jg];
        if (jg > qg) v = -3.0e38f;  // causal
        sv[tn][r] = v;
      }
    }

    // online softmax (row = r + 8*(lane>=16); columns across 16 lanes)
#pragma unroll
    for (int r = 0; r < 8; ++r) {
      float mx = fmaxf(fmaxf(sv[0][r], sv[1][r]), fmaxf(sv[2][r], sv[3][r]));
#pragma unroll
      for (int off = 8; off >= 1; off >>= 1)
        mx = fmaxf(mx, __shfl_xor(mx, off, 32));
      float mnew    = fmaxf(mrow[r], mx);
      float rescale = __expf(mrow[r] - mnew);
      float rs = 0.0f;
#pragma unroll
      for (int tn = 0; tn < 4; ++tn) {
        float p = __expf(sv[tn][r] - mnew);
        sv[tn][r] = p;
        rs += p;
      }
#pragma unroll
      for (int off = 8; off >= 1; off >>= 1) rs += __shfl_xor(rs, off, 32);
      lrow[r] = lrow[r] * rescale + rs;
      mrow[r] = mnew;
#pragma unroll
      for (int td = 0; td < 4; ++td) aO[td][r] *= rescale;
    }

    // C-layout -> A-layout for P via LDS (same-wave RAW: explicit dscnt wait)
#pragma unroll
    for (int tn = 0; tn < 4; ++tn)
#pragma unroll
      for (int r = 0; r < 8; ++r)
        Ps[wave * 16 + r + rh][tn * 16 + cl] = (_Float16)sv[tn][r];
    asm volatile("s_wait_dscnt 0" ::: "memory");
    v16h pa0 = ld_afrag(&Ps[wave * 16][0], 72);   // K = j 0..31
    v16h pa1 = ld_afrag(&Ps[wave * 16][32], 72);  // K = j 32..63

    // O stripe += P @ V  -- hoisted fragments, then wmma burst
    v16h vb0[4], vb1[4];
#pragma unroll
    for (int td = 0; td < 4; ++td) {
      vb0[td] = ld_bfrag(&Vt[td * 16][0], 72);
      vb1[td] = ld_bfrag(&Vt[td * 16][32], 72);
    }
#pragma unroll
    for (int td = 0; td < 4; ++td) {
      aO[td] = __builtin_amdgcn_wmma_f32_16x16x32_f16(false, pa0, false, vb0[td],
                                                      (short)0, aO[td], false, false);
      aO[td] = __builtin_amdgcn_wmma_f32_16x16x32_f16(false, pa1, false, vb1[td],
                                                      (short)0, aO[td], false, false);
    }
    __syncthreads();
  }

  // epilogue: O /= l, write f16 into [B, T, D] with d = h*64 + dk
#pragma unroll
  for (int td = 0; td < 4; ++td) {
    int d = h * DK_ + td * 16 + cl;
#pragma unroll
    for (int r = 0; r < 8; ++r) {
      int qg = q0 + wave * 16 + r + rh;
      Oh[((size_t)(b * T_ + qg)) * D_ + d] = (_Float16)(aO[td][r] / lrow[r]);
    }
  }
}

// ---------------------------------------------------------------------------
// Host launcher
// ---------------------------------------------------------------------------
extern "C" void kernel_launch(void* const* d_in, const int* in_sizes, int n_in,
                              void* d_out, int out_size, void* d_ws,
                              size_t ws_size, hipStream_t stream) {
  (void)in_sizes; (void)n_in; (void)out_size; (void)ws_size;
  const float* x     = (const float*)d_in[0];
  const float* dist  = (const float*)d_in[1];
  /* d_in[2] = mask: causal, computed analytically in-kernel */
  const float* Wq = (const float*)d_in[3];
  const float* bq = (const float*)d_in[4];
  const float* Wk = (const float*)d_in[5];
  const float* bk = (const float*)d_in[6];
  const float* Wv = (const float*)d_in[7];
  const float* bv = (const float*)d_in[8];
  const float* Wo = (const float*)d_in[9];
  const float* bo = (const float*)d_in[10];
  const float* alpha = (const float*)d_in[11];

  char* ws = (char*)d_ws;
  _Float16* Xh  = (_Float16*)(ws);                               // 8 MB
  _Float16* Wqh = (_Float16*)(ws + (size_t)(8)  * (1u << 20));
  _Float16* Wkh = (_Float16*)(ws + (size_t)(10) * (1u << 20));
  _Float16* Wvh = (_Float16*)(ws + (size_t)(12) * (1u << 20));
  _Float16* Woh = (_Float16*)(ws + (size_t)(14) * (1u << 20));
  _Float16* Qh  = (_Float16*)(ws + (size_t)(16) * (1u << 20));   // [B,H,T,DK]
  _Float16* Kh  = (_Float16*)(ws + (size_t)(24) * (1u << 20));
  _Float16* Vh  = (_Float16*)(ws + (size_t)(32) * (1u << 20));
  _Float16* Oh  = (_Float16*)(ws + (size_t)(40) * (1u << 20));   // [B,T,D]

  const int NX = B_ * T_ * D_;
  const int NW = D_ * D_;
  f32_to_f16_kernel<<<(NX + 255) / 256, 256, 0, stream>>>(x,  Xh,  NX);
  f32_to_f16_kernel<<<(NW + 255) / 256, 256, 0, stream>>>(Wq, Wqh, NW);
  f32_to_f16_kernel<<<(NW + 255) / 256, 256, 0, stream>>>(Wk, Wkh, NW);
  f32_to_f16_kernel<<<(NW + 255) / 256, 256, 0, stream>>>(Wv, Wvh, NW);
  f32_to_f16_kernel<<<(NW + 255) / 256, 256, 0, stream>>>(Wo, Woh, NW);

  dim3 gg(D_ / 64, (B_ * T_) / 128);
  gemm128x64<1><<<gg, 256, 0, stream>>>(Xh, Wqh, bq, Qh, 0.125f);  // 1/sqrt(64)
  gemm128x64<1><<<gg, 256, 0, stream>>>(Xh, Wkh, bk, Kh, 1.0f);
  gemm128x64<1><<<gg, 256, 0, stream>>>(Xh, Wvh, bv, Vh, 1.0f);

  attn_flash_kernel<<<dim3(B_ * H_, T_ / 64), 128, 0, stream>>>(
      Qh, Kh, Vh, dist, alpha, Oh);

  gemm128x64<0><<<gg, 256, 0, stream>>>(Oh, Woh, bo, d_out, 1.0f);
}
